// FieldAttentionModule_85487029059844
// MI455X (gfx1250) — compile-verified, matrix operand order checked by reading, not run on hardware
//
#include <hip/hip_runtime.h>

typedef _Float16 half_t;
typedef __attribute__((ext_vector_type(2)))  _Float16 h2v;
typedef __attribute__((ext_vector_type(16))) _Float16 v16h;
typedef __attribute__((ext_vector_type(8)))  float    v8f;

#define WMMA_F16(a, b, c) \
  __builtin_amdgcn_wmma_f32_16x16x32_f16(false, (a), false, (b), (short)0, (c), false, false)

namespace {
constexpr int S = 64, LF = 128, ATT = 128, RANK = 32, HC = 32, NH = 4;
constexpr int NWIN = 2 * 1024;
constexpr int NTHREADS = 256;           // 8 waves

// fused-weight workspace layout (half_t elements); each matrix stored in
// WMMA B-fragment-swizzled order: tile(kt,nt) -> 32 lanes x 16 halves.
constexpr int WS_WQ  = 0;
constexpr int WS_WKV = WS_WQ  + LF * ATT;       // 16384
constexpr int WS_WO  = WS_WKV + LF * 2 * ATT;   // 49152
constexpr int WS_W1  = WS_WO  + ATT * LF;       // 65536
constexpr int WS_W2  = WS_W1  + LF * ATT;       // 81920
constexpr int WS_TOT = WS_W2  + ATT * LF;       // 98304 halfs = 192 KB

// LDS: sH(=o,h2) + sQ(=gelu) + sK + sVt (4 x 16KB f16)
//      + scores f32 [4][64][64] (probs f16 aliased into its lower half)
//      + x/x1 f32 [64][128]
// = 64K + 64K + 32K = 163840 B exactly -> 2 blocks per 320KB WGP.
constexpr size_t SMEM_BYTES =
    (size_t)4 * S * LF * 2 + (size_t)NH * S * S * 4 + (size_t)S * LF * 4;
}  // namespace

// ------------- fragment loaders (CDNA5 wave32 WMMA layouts, vectorized) -------

// Per-lane data = two contiguous 16-byte chunks at row (m0 + lane&15):
// chunk0 = K[0..7], chunk1 = K[16..23] (lanes >= 16 shifted by 8 in K).
// Serves A fragments and B^T fragments.
__device__ __forceinline__ v16h load_frag_rows(const half_t* m, int ld, int m0,
                                               int k0, int lane) {
  const half_t* p = m + (size_t)(m0 + (lane & 15)) * ld + k0 + ((lane & 16) ? 8 : 0);
  union { uint4 u[2]; v16h h; } u;
  u.u[0] = *(const uint4*)(p);
  u.u[1] = *(const uint4*)(p + 16);
  return u.h;
}

// Pre-swizzled weight B fragment: one 32-byte contiguous load per lane.
__device__ __forceinline__ v16h load_Bw(const half_t* w, int NT, int kk, int n0,
                                        int lane) {
  const half_t* p = w + ((((size_t)(kk >> 5) * NT + (n0 >> 4)) * 32 + lane) << 4);
  union { uint4 u[2]; v16h h; } u;
  u.u[0] = *(const uint4*)(p);
  u.u[1] = *(const uint4*)(p + 16);
  return u.h;
}

// Decode swizzled fragment index -> (k, n) for the fuse kernel.
__device__ __forceinline__ void frag_decode(int idx, int NT, int& k, int& n) {
  int tile   = idx >> 9;          // 512 halfs per 32x16 tile
  int within = idx & 511;
  int lane   = within >> 4;
  int e      = within & 15;
  int kt = tile / NT, nt = tile - kt * NT;
  int kb = (lane & 16) ? 8 : 0;
  int j = e >> 1, lo = e & 1;
  k = kt * 32 + kb + 2 * j + ((j >= 4) ? 8 : 0) + lo;
  n = nt * 16 + (lane & 15);
}

// 4-lane (one token) sum reduction via wave32 lane shuffles.
__device__ __forceinline__ float red4(float v) {
  v += __shfl_xor(v, 1, 32);
  v += __shfl_xor(v, 2, 32);
  return v;
}

// ---------------- weight fusion: (U@V) -> dense f16 fragments ------------------

__global__ __launch_bounds__(256) void fam_fuse_weights(
    const float* __restrict__ Wq_u, const float* __restrict__ Wq_v,
    const float* __restrict__ Wkv_u, const float* __restrict__ Wkv_v,
    const float* __restrict__ Wo_u, const float* __restrict__ Wo_v,
    const float* __restrict__ W1, const float* __restrict__ W2,
    half_t* __restrict__ ws) {
  int i = blockIdx.x * blockDim.x + threadIdx.x;
  if (i >= WS_TOT) return;
  int k, n;
  float val;
  if (i < WS_WKV) {                             // Wq fused, 1/sqrt(HC) folded in
    frag_decode(i - WS_WQ, ATT / 16, k, n);
    float acc = 0.f;
#pragma unroll
    for (int r = 0; r < RANK; ++r) acc += Wq_u[k * RANK + r] * Wq_v[r * ATT + n];
    val = acc * 0.17677669529663687f;           // 1/sqrt(32)
  } else if (i < WS_WO) {                       // Wkv fused
    frag_decode(i - WS_WKV, 2 * ATT / 16, k, n);
    float acc = 0.f;
#pragma unroll
    for (int r = 0; r < RANK; ++r) acc += Wkv_u[k * RANK + r] * Wkv_v[r * 2 * ATT + n];
    val = acc;
  } else if (i < WS_W1) {                       // Wo fused
    frag_decode(i - WS_WO, LF / 16, k, n);
    float acc = 0.f;
#pragma unroll
    for (int r = 0; r < RANK; ++r) acc += Wo_u[k * RANK + r] * Wo_v[r * LF + n];
    val = acc;
  } else if (i < WS_W2) {                       // W1 -> f16
    frag_decode(i - WS_W1, ATT / 16, k, n);
    val = W1[k * ATT + n];
  } else {                                      // W2 -> f16
    frag_decode(i - WS_W2, LF / 16, k, n);
    val = W2[k * LF + n];
  }
  ws[i] = (half_t)val;
}

// ---------------- fused attention block: 1 workgroup per (b,g) window ----------

__global__ __launch_bounds__(NTHREADS) void fam_main(
    const float* __restrict__ x, const half_t* __restrict__ ws,
    const float* __restrict__ kv_b, const float* __restrict__ gamma,
    const float* __restrict__ b1, const float* __restrict__ b2,
    const float* __restrict__ gamma_mlp, float* __restrict__ out) {
  extern __shared__ char smem[];
  half_t* sH  = (half_t*)smem;              // [64][128] h -> o -> h2 (f16)
  half_t* sQ  = sH + S * LF;                // [64][128] q -> gelu(g)
  half_t* sK  = sQ + S * LF;                // [64][128] k (row-major)
  half_t* sVt = sK + S * LF;                // [128][64] v transposed
  float*  sS  = (float*)(sVt + S * LF);     // [4][64][64] scores f32; probs f16
                                            //   aliased into lower half per head
  float*  sX  = (float*)(sS + NH * S * S);  // [64][128] x, then x1 (f32)

  const int tid  = threadIdx.x;
  const int lane = tid & 31;
  const int wid  = tid >> 5;                // 8 waves
  const int win  = blockIdx.x;              // 2048 windows
  const int moff = (lane & 16) ? 8 : 0;     // C/D row offset per lane group
  const int ncol = lane & 15;               // C/D column per lane
  const float* xw = x + (size_t)win * S * LF;

  const int tok  = tid >> 2;                // LN: 4 lanes per token
  const int part = tid & 3;

  // ---- Phase A0: coalesced stream of x into sX (float4)
  {
    const float4* xg = (const float4*)xw;
    float4* xs4 = (float4*)sX;
#pragma unroll
    for (int i = tid; i < S * LF / 4; i += NTHREADS) xs4[i] = xg[i];
  }
  __syncthreads();

  // ---- Phase A1: LayerNorm(x) -> sH (f16); 4 lanes per token, shfl reduce
  {
    const float* xs = sX + tok * LF + part * 32;
    float sum = 0.f, ssq = 0.f;
#pragma unroll
    for (int j = 0; j < 8; ++j) {
      float4 v = ((const float4*)xs)[j];
      sum += v.x + v.y + v.z + v.w;
      ssq += v.x * v.x + v.y * v.y + v.z * v.z + v.w * v.w;
    }
    sum = red4(sum); ssq = red4(ssq);
    float mean = sum * (1.f / LF);
    float var  = ssq * (1.f / LF) - mean * mean;
    float rs   = rsqrtf(var + 1e-5f);
    h2v* hr = (h2v*)(sH + tok * LF + part * 32);
#pragma unroll
    for (int j = 0; j < 16; ++j)
      hr[j] = (h2v){(half_t)((xs[2 * j] - mean) * rs),
                    (half_t)((xs[2 * j + 1] - mean) * rs)};
  }
  __syncthreads();

  // ---- Phase B: q = h@Wq (pre-scaled), [k|v] = h@Wkv + kv_b
  for (int t = wid; t < 96; t += 8) {
    int mr = t & 3, ct = t >> 2;
    int m0 = mr * 16;
    v8f acc = {};
    if (ct < 8) {
      int n0 = ct * 16;
#pragma unroll
      for (int kk = 0; kk < LF; kk += 32)
        acc = WMMA_F16(load_frag_rows(sH, LF, m0, kk, lane),
                       load_Bw(ws + WS_WQ, ATT / 16, kk, n0, lane), acc);
#pragma unroll
      for (int r = 0; r < 8; ++r)
        sQ[(size_t)(m0 + moff + r) * LF + n0 + ncol] = (half_t)acc[r];
    } else {
      int n0 = (ct - 8) * 16;               // 0..240 within [k|v]
#pragma unroll
      for (int kk = 0; kk < LF; kk += 32)
        acc = WMMA_F16(load_frag_rows(sH, LF, m0, kk, lane),
                       load_Bw(ws + WS_WKV, 2 * ATT / 16, kk, n0, lane), acc);
      float bias = kv_b[n0 + ncol];
      if (n0 < ATT) {                        // k: row-major
#pragma unroll
        for (int r = 0; r < 8; ++r)
          sK[(size_t)(m0 + moff + r) * LF + n0 + ncol] = (half_t)(acc[r] + bias);
      } else {                               // v: transposed, packed b128 store
        union { uint4 u; half_t h[8]; } st;
#pragma unroll
        for (int r = 0; r < 8; ++r) st.h[r] = (half_t)(acc[r] + bias);
        *(uint4*)(sVt + (size_t)(n0 - ATT + ncol) * S + m0 + moff) = st.u;
      }
    }
  }
  __syncthreads();

  // ---- Phase C: scores = q_h @ k_h^T; 2 waves per head, 8 tiles each
  {
    const int head = wid >> 1, sub = wid & 1;
    float* sSh = sS + head * S * S;
#pragma unroll
    for (int ij = sub * 8; ij < sub * 8 + 8; ++ij) {
      int i = ij >> 2, j = ij & 3;
      v8f acc = {};
      acc = WMMA_F16(load_frag_rows(sQ, LF, i * 16, head * HC, lane),
                     load_frag_rows(sK, LF, j * 16, head * HC, lane), acc);
#pragma unroll
      for (int r = 0; r < 8; ++r)
        sSh[(i * 16 + moff + r) * S + j * 16 + ncol] = acc[r];
    }
  }
  __syncthreads();

  // ---- softmax: one row per lane; probs (f16) aliased into score buffer.
  // Lane rr's f16 probs land in the bytes of score row rr/2, not its own row.
  {
    const int head = wid >> 1;
    const int rr   = (wid & 1) * 32 + lane;
    const float* row = sS + head * S * S + rr * S;
    const float4* row4 = (const float4*)row;
    half_t* prow = (half_t*)(sS + head * S * S) + rr * S;
    float mx = -3.4e38f;
#pragma unroll
    for (int c = 0; c < S / 4; ++c) {
      float4 v = row4[c];
      mx = fmaxf(mx, fmaxf(fmaxf(v.x, v.y), fmaxf(v.z, v.w)));
    }
    float sum = 0.f;
#pragma unroll
    for (int c = 0; c < S / 4; ++c) {
      float4 v = row4[c];
      sum += __expf(v.x - mx) + __expf(v.y - mx) + __expf(v.z - mx) + __expf(v.w - mx);
    }
    float inv = 1.f / sum;
#pragma unroll
    for (int c = 0; c < S / 4; ++c) {
      float4 v = row4[c];
      union { uint2 u; half_t h[4]; } pk;
      pk.h[0] = (half_t)(__expf(v.x - mx) * inv);
      pk.h[1] = (half_t)(__expf(v.y - mx) * inv);
      pk.h[2] = (half_t)(__expf(v.z - mx) * inv);
      pk.h[3] = (half_t)(__expf(v.w - mx) * inv);
      *(uint2*)(prow + 4 * c) = pk.u;
    }
  }
  __syncthreads();

  // ---- Phase D: o_h = P_h @ v_h -> sH (h is dead); 2 waves per head
  {
    const int head = wid >> 1, sub = wid & 1;
    const half_t* sPh = (const half_t*)(sS + head * S * S);
#pragma unroll
    for (int t8 = sub * 4; t8 < sub * 4 + 4; ++t8) {
      int i = t8 >> 1, jc = t8 & 1;
      int m0 = i * 16, n0 = head * HC + jc * 16;
      v8f acc = {};
#pragma unroll
      for (int ks = 0; ks < 2; ++ks)
        acc = WMMA_F16(load_frag_rows(sPh, S, m0, ks * 32, lane),
                       load_frag_rows(sVt, S, n0, ks * 32, lane), acc);
#pragma unroll
      for (int r = 0; r < 8; ++r)
        sH[(size_t)(m0 + moff + r) * LF + n0 + ncol] = (half_t)acc[r];
    }
  }
  __syncthreads();

  // ---- Phase E: x1 = x + gamma * (o @ Wo)   (sX updated in place)
  for (int t = wid; t < 32; t += 8) {
    int mr = t >> 3, nc = t & 7;
    int m0 = mr * 16, n0 = nc * 16;
    v8f acc = {};
#pragma unroll
    for (int kk = 0; kk < LF; kk += 32)
      acc = WMMA_F16(load_frag_rows(sH, LF, m0, kk, lane),
                     load_Bw(ws + WS_WO, LF / 16, kk, n0, lane), acc);
    float g = gamma[n0 + ncol];
#pragma unroll
    for (int r = 0; r < 8; ++r) {
      size_t idx = (size_t)(m0 + moff + r) * LF + n0 + ncol;
      sX[idx] += g * acc[r];
    }
  }
  __syncthreads();

  // ---- Phase F: LayerNorm(x1) -> sH (o is dead)
  {
    const float* xs = sX + tok * LF + part * 32;
    float sum = 0.f, ssq = 0.f;
#pragma unroll
    for (int j = 0; j < 8; ++j) {
      float4 v = ((const float4*)xs)[j];
      sum += v.x + v.y + v.z + v.w;
      ssq += v.x * v.x + v.y * v.y + v.z * v.z + v.w * v.w;
    }
    sum = red4(sum); ssq = red4(ssq);
    float mean = sum * (1.f / LF);
    float var  = ssq * (1.f / LF) - mean * mean;
    float rs   = rsqrtf(var + 1e-5f);
    h2v* hr = (h2v*)(sH + tok * LF + part * 32);
#pragma unroll
    for (int j = 0; j < 16; ++j)
      hr[j] = (h2v){(half_t)((xs[2 * j] - mean) * rs),
                    (half_t)((xs[2 * j + 1] - mean) * rs)};
  }
  __syncthreads();

  // ---- Phase G: g = gelu(h2 @ W1 + b1) -> sQ (q is dead)
  for (int t = wid; t < 32; t += 8) {
    int mr = t >> 3, nc = t & 7;
    int m0 = mr * 16, n0 = nc * 16;
    v8f acc = {};
#pragma unroll
    for (int kk = 0; kk < LF; kk += 32)
      acc = WMMA_F16(load_frag_rows(sH, LF, m0, kk, lane),
                     load_Bw(ws + WS_W1, ATT / 16, kk, n0, lane), acc);
    float bb = b1[n0 + ncol];
#pragma unroll
    for (int r = 0; r < 8; ++r) {
      float v  = acc[r] + bb;
      float u  = 0.7978845608028654f * (v + 0.044715f * v * v * v);
      float ge = 0.5f * v * (1.f + tanhf(u));   // tanh-approx gelu (jax default)
      sQ[(size_t)(m0 + moff + r) * LF + n0 + ncol] = (half_t)ge;
    }
  }
  __syncthreads();

  // ---- Phase H: out = x1 + gamma_mlp * (g @ W2 + b2)
  const float gm = gamma_mlp[0];
  float* ow = out + (size_t)win * S * LF;
  for (int t = wid; t < 32; t += 8) {
    int mr = t >> 3, nc = t & 7;
    int m0 = mr * 16, n0 = nc * 16;
    v8f acc = {};
#pragma unroll
    for (int kk = 0; kk < LF; kk += 32)
      acc = WMMA_F16(load_frag_rows(sQ, LF, m0, kk, lane),
                     load_Bw(ws + WS_W2, LF / 16, kk, n0, lane), acc);
    float bb = b2[n0 + ncol];
#pragma unroll
    for (int r = 0; r < 8; ++r) {
      size_t idx = (size_t)(m0 + moff + r) * LF + n0 + ncol;
      ow[idx] = sX[idx] + gm * (acc[r] + bb);
    }
  }
}

// ---------------- launcher ----------------

extern "C" void kernel_launch(void* const* d_in, const int* in_sizes, int n_in,
                              void* d_out, int out_size, void* d_ws, size_t ws_size,
                              hipStream_t stream) {
  const float* x         = (const float*)d_in[0];
  const float* Wq_u      = (const float*)d_in[1];
  const float* Wq_v      = (const float*)d_in[2];
  const float* Wkv_u     = (const float*)d_in[3];
  const float* Wkv_v     = (const float*)d_in[4];
  const float* kv_b      = (const float*)d_in[5];
  const float* Wo_u      = (const float*)d_in[6];
  const float* Wo_v      = (const float*)d_in[7];
  const float* gamma     = (const float*)d_in[8];
  const float* W1        = (const float*)d_in[9];
  const float* b1        = (const float*)d_in[10];
  const float* W2        = (const float*)d_in[11];
  const float* b2        = (const float*)d_in[12];
  const float* gamma_mlp = (const float*)d_in[13];
  (void)in_sizes; (void)n_in; (void)out_size; (void)ws_size;

  half_t* ws = (half_t*)d_ws;   // needs WS_TOT*2 = 192 KB of workspace

  // allow >64KB dynamic LDS (160KB/block -> 2 blocks per 320KB WGP)
  (void)hipFuncSetAttribute((const void*)fam_main,
                            hipFuncAttributeMaxDynamicSharedMemorySize,
                            (int)SMEM_BYTES);

  fam_fuse_weights<<<(WS_TOT + 255) / 256, 256, 0, stream>>>(
      Wq_u, Wq_v, Wkv_u, Wkv_v, Wo_u, Wo_v, W1, W2, ws);

  fam_main<<<NWIN, NTHREADS, SMEM_BYTES, stream>>>(
      x, ws, kv_b, gamma, b1, b2, gamma_mlp, (float*)d_out);
}
